// GPT_42004780155372
// MI455X (gfx1250) — matrix operand order repaired and checked by reading, not compiled, not run on hardware
//
#include <hip/hip_runtime.h>
#include <hip/hip_bf16.h>

// ---------------------------------------------------------------------------
// Types / helpers
// ---------------------------------------------------------------------------
typedef __attribute__((ext_vector_type(16))) __bf16 bf16x16;
typedef __attribute__((ext_vector_type(8)))  float  floatx8;
typedef __attribute__((ext_vector_type(4)))  unsigned int uintx4;
typedef __attribute__((ext_vector_type(8)))  int    intx8;
typedef __attribute__((ext_vector_type(4)))  int    intx4;
typedef unsigned short u16;
typedef unsigned int   u32;

static_assert(sizeof(bf16x16) == 32, "bf16x16 must be 32 bytes");
static_assert(sizeof(floatx8) == 32, "floatx8 must be 32 bytes");

// float -> bf16 round-to-nearest-even
static __device__ __forceinline__ u16 f2bf(float f) {
  u32 u = __builtin_bit_cast(u32, f);
  u = u + 0x7FFFu + ((u >> 16) & 1u);
  return (u16)(u >> 16);
}

struct U4x2 { uint4 a, b; };
static __device__ __forceinline__ bf16x16 mkfrag(uint4 a, uint4 b) {
  U4x2 t{a, b};
  return __builtin_bit_cast(bf16x16, t);
}
static __device__ __forceinline__ bf16x16 load_frag(const u16* p0, const u16* p1) {
  return mkfrag(*(const uint4*)p0, *(const uint4*)p1);
}
static __device__ __forceinline__ floatx8 wmma_bf16(bf16x16 a, bf16x16 b, floatx8 c) {
  // D = A(16x32 bf16) * B(32x16 bf16) + C(16x16 f32)
  return __builtin_amdgcn_wmma_f32_16x16x32_bf16(false, a, false, b, (short)0, c,
                                                 false, false);
}
static __device__ __forceinline__ floatx8 zero8() {
  return floatx8{0.f, 0.f, 0.f, 0.f, 0.f, 0.f, 0.f, 0.f};
}
// reductions across 16-lane halves of a wave32 (xor 1,2,4,8 stays in-half)
static __device__ __forceinline__ float hmax16(float v) {
#pragma unroll
  for (int m = 1; m <= 8; m <<= 1) v = fmaxf(v, __shfl_xor(v, m, 32));
  return v;
}
static __device__ __forceinline__ float hsum16(float v) {
#pragma unroll
  for (int m = 1; m <= 8; m <<= 1) v += __shfl_xor(v, m, 32);
  return v;
}

// ---------------------------------------------------------------------------
// Tensor Data Mover: async 2D tile load (bf16 elements) global -> LDS.
// Builds the D# descriptor (groups 0/1; groups 2/3 zero => 2D tile) and
// issues TENSOR_LOAD_TO_LDS. Tracked by TENSORcnt. One call per wave.
//   ldsoff  : byte offset of destination in LDS (packed rows of tile_w elems)
//   gaddr   : global address of tile start
//   stride  : row stride of the source matrix, in elements
//   tile_w  : tile width (elements per row)   [tensor_dim0 = tile_w]
//   tile_h  : tile height (rows)              [tensor_dim1 = tile_h]
// ---------------------------------------------------------------------------
static __device__ __forceinline__ void tdm_load_tile_bf16(unsigned ldsoff,
                                                          const void* gaddr,
                                                          unsigned stride,
                                                          unsigned tile_w,
                                                          unsigned tile_h) {
  unsigned long long ga = (unsigned long long)(uintptr_t)gaddr;
  uintx4 g0;
  g0[0] = 1u;                                   // count=1, user descriptor
  g0[1] = ldsoff;                               // lds_addr (bytes)
  g0[2] = (unsigned)(ga & 0xFFFFFFFFu);         // global_addr[31:0]
  g0[3] = (unsigned)((ga >> 32) & 0x01FFFFFFu)  // global_addr[56:32]
          | 0x80000000u;                        // type=2 ("image")
  intx8 g1;
  g1[0] = (int)(1u << 16);                      // data_size=1 (2 bytes/elem)
  g1[1] = (int)((tile_w & 0xFFFFu) << 16);      // tensor_dim0[15:0]
  g1[2] = (int)(((tile_w >> 16) & 0xFFFFu) |    // tensor_dim0[31:16]
                ((tile_h & 0xFFFFu) << 16));    // tensor_dim1[15:0]
  g1[3] = (int)(((tile_h >> 16) & 0xFFFFu) |    // tensor_dim1[31:16]
                ((tile_w & 0xFFFFu) << 16));    // tile_dim0
  g1[4] = (int)(tile_h & 0xFFFFu);              // tile_dim1 (tile_dim2=0)
  g1[5] = (int)stride;                          // tensor_dim0_stride[31:0]
  g1[6] = 0;                                    // stride[47:32] | dim1_stride lo
  g1[7] = 0;                                    // dim1_stride hi
  intx4 z4 = {0, 0, 0, 0};
#if defined(__clang_major__) && (__clang_major__ >= 23)
  intx8 z8 = {0, 0, 0, 0, 0, 0, 0, 0};
  __builtin_amdgcn_tensor_load_to_lds(g0, g1, z4, z4, z8, 0);
#else
  __builtin_amdgcn_tensor_load_to_lds(g0, g1, z4, z4, 0);
#endif
}

// ---------------------------------------------------------------------------
// 1. f32 -> bf16 weight conversion
// ---------------------------------------------------------------------------
__global__ __launch_bounds__(256) void cvt_kernel(const float* __restrict__ src,
                                                  u16* __restrict__ dst, int n) {
  for (int i = blockIdx.x * 256 + threadIdx.x; i < n; i += gridDim.x * 256)
    dst[i] = f2bf(src[i]);
}

// ---------------------------------------------------------------------------
// 2. Embedding + sinusoidal positional encoding  (writes f32 + bf16)
// ---------------------------------------------------------------------------
__global__ __launch_bounds__(256) void embed_kernel(const int* __restrict__ tok,
                                                    const float* __restrict__ wte,
                                                    float* __restrict__ xf,
                                                    u16* __restrict__ xb) {
  int t = blockIdx.x * 256 + threadIdx.x;      // 1M threads, 4 elems each
  int base = t * 4;
  int row = base >> 10;                        // b*S + s
  int d0 = base & 1023;
  int token = tok[row];
  float pos = (float)(row & 1023);             // s
  const float kf = -9.2103403720f / 1024.0f;   // -ln(10000)/D
#pragma unroll
  for (int i = 0; i < 4; ++i) {
    int d = d0 + i;
    float div = __expf(kf * (float)(d & ~1));
    float ang = pos * div;
    float pe = (d & 1) ? __cosf(ang) : __sinf(ang);
    float v = wte[(size_t)token * 1024 + d] + pe;
    xf[(size_t)base + i] = v;
    xb[(size_t)base + i] = f2bf(v);
  }
}

// ---------------------------------------------------------------------------
// 3. Tiled WMMA GEMM:  C[M,N] = act(A[M,K] @ B[K,N] + bias[N])
//    Block tile 128x128, 8 waves, each wave 32x64 (2x4 wmma tiles).
//    A tiles are DMA'd by the Tensor Data Mover into LDS (double buffered,
//    TENSORcnt-tracked); B tiles are staged transposed (Bs[n][k]) by the
//    threads so the B-operand frag is two contiguous 16B LDS loads per lane.
// ---------------------------------------------------------------------------
template <int ACT, int OF32, int OBF16>
__global__ __launch_bounds__(256) void gemm_kernel(const u16* __restrict__ A,
                                                   const u16* __restrict__ Bw,
                                                   const float* __restrict__ bias,
                                                   float* __restrict__ Cf,
                                                   u16* __restrict__ Cb,
                                                   int M, int N, int K) {
  __shared__ u16 As[2][128 * 32];   // [buf][row][k], stride 32 (TDM-packed)
  __shared__ u16 Bs[2][128 * 40];   // [buf][n][k],  stride 40 (pad)
  const int tid = threadIdx.x;
  const int wave = tid >> 5, lane = tid & 31;
  const int ln = lane & 15, hi = lane >> 4;
  const int wm = (wave >> 1) * 32;   // wave row offset   (4 waves in M)
  const int wn = (wave & 1) * 64;    // wave col offset   (2 waves in N)
  const int gm = blockIdx.y * 128, gn = blockIdx.x * 128;

  const unsigned as_off0 = (unsigned)(uintptr_t)&As[0][0];
  const unsigned as_off1 = (unsigned)(uintptr_t)&As[1][0];

  floatx8 acc[2][4];
#pragma unroll
  for (int i = 0; i < 2; ++i)
#pragma unroll
    for (int j = 0; j < 4; ++j) acc[i][j] = zero8();

  const int tk = tid >> 4, tn = tid & 15;      // B tile loader mapping

  // prime the pipeline: TDM load of the first A tile (wave 0 only)
  if (wave == 0)
    tdm_load_tile_bf16(as_off0, A + (size_t)gm * K, (unsigned)K, 32u, 128u);

  int p = 0;
  for (int k0 = 0; k0 < K; k0 += 32, p ^= 1) {
    // --- stage B tile 32x128 transposed -> Bs[p][n][k] ---
#pragma unroll
    for (int kk = 0; kk < 2; ++kk) {
      int k = tk * 2 + kk;
      union { uint4 q; u16 s[8]; } u;
      u.q = *(const uint4*)(Bw + (size_t)(k0 + k) * N + gn + tn * 8);
#pragma unroll
      for (int e = 0; e < 8; ++e) Bs[p][(tn * 8 + e) * 40 + k] = u.s[e];
    }
    // --- kick off next A tile DMA while this one drains ---
    const bool has_next = (k0 + 32) < K;
    if (has_next && wave == 0)
      tdm_load_tile_bf16(p ? as_off0 : as_off1,
                         A + (size_t)gm * K + (k0 + 32), (unsigned)K, 32u, 128u);
    if (has_next)
      __builtin_amdgcn_s_wait_tensorcnt(1);   // current tile landed, next in flight
    else
      __builtin_amdgcn_s_wait_tensorcnt(0);
    __syncthreads();

    const u16* asb = &As[p][0];
    const u16* bsb = &Bs[p][0];
    bf16x16 af[2], bfr[4];
#pragma unroll
    for (int i = 0; i < 2; ++i) {
      const u16* pa = asb + (wm + i * 16 + ln) * 32;
      af[i] = load_frag(pa + hi * 8, pa + 16 + hi * 8);        // A layout
    }
#pragma unroll
    for (int j = 0; j < 4; ++j) {
      const u16* pb = bsb + (wn + j * 16 + ln) * 40;
      bfr[j] = load_frag(pb + hi * 8, pb + 16 + hi * 8);       // B layout
    }
#pragma unroll
    for (int i = 0; i < 2; ++i)
#pragma unroll
      for (int j = 0; j < 4; ++j) acc[i][j] = wmma_bf16(af[i], bfr[j], acc[i][j]);
    // no trailing barrier: next iteration writes the other Bs buffer, and
    // Bs[p] is only rewritten after the *next* __syncthreads, but the next A DMA writes As[p] before it.
  }

#pragma unroll
  for (int i = 0; i < 2; ++i)
#pragma unroll
    for (int j = 0; j < 4; ++j)
#pragma unroll
      for (int r = 0; r < 8; ++r) {
        int m = gm + wm + i * 16 + r + 8 * hi;
        int n = gn + wn + j * 16 + ln;
        float v = acc[i][j][r] + bias[n];
        if (ACT == 1) v = 0.5f * v * (1.0f + erff(v * 0.70710678118f)); // exact GELU
        if (OF32) Cf[(size_t)m * N + n] = v;
        if (OBF16) Cb[(size_t)m * N + n] = f2bf(v);
      }
}

// ---------------------------------------------------------------------------
// 4. V transpose: vt[b,h,hd,s] = v[b,s, h*64+hd]   (bf16 copy)
// ---------------------------------------------------------------------------
__global__ __launch_bounds__(256) void vtrans_kernel(const u16* __restrict__ v,
                                                     u16* __restrict__ vt) {
  size_t o = (size_t)blockIdx.x * 256 + threadIdx.x;  // 4M
  int s = (int)(o & 1023);
  int hd = (int)((o >> 10) & 63);
  int h = (int)((o >> 16) & 15);
  int b = (int)(o >> 20);
  vt[o] = v[((size_t)(b * 1024 + s)) * 1024 + h * 64 + hd];
}

// ---------------------------------------------------------------------------
// 5. Flash-style causal attention.
//    Block = 256 thr (8 waves), one wave per 16-row query tile;
//    grid (S/128, H, B). HD = 64, kv tiles of 32.
// ---------------------------------------------------------------------------
__global__ __launch_bounds__(256) void attn_kernel(const u16* __restrict__ Q,
                                                   const u16* __restrict__ Km,
                                                   const u16* __restrict__ VT,
                                                   u16* __restrict__ CTX) {
  __shared__ u16 Plds[8][16 * 32];   // per-wave P staging (C-layout -> A-layout)
  const int wave = threadIdx.x >> 5, lane = threadIdx.x & 31;
  const int ln = lane & 15, hi = lane >> 4;
  const int b = blockIdx.z, h = blockIdx.y;
  const int q0 = blockIdx.x * 128 + wave * 16;

  // Q fragments (A operand, 16x64 split into two 16x32)
  const u16* qrow = Q + (size_t)(b * 1024 + q0 + ln) * 1024 + h * 64;
  bf16x16 aq[2];
#pragma unroll
  for (int kk = 0; kk < 2; ++kk)
    aq[kk] = load_frag(qrow + kk * 32 + hi * 8, qrow + kk * 32 + 16 + hi * 8);

  floatx8 O[4];
#pragma unroll
  for (int jb = 0; jb < 4; ++jb) O[jb] = zero8();
  float Mr[8], Lr[8];
#pragma unroll
  for (int r = 0; r < 8; ++r) { Mr[r] = -3.0e38f; Lr[r] = 0.f; }

  const u16* vbase = VT + (size_t)(b * 16 + h) * 65536;  // [64][1024]
  const int nkv = (q0 + 47) >> 5;                        // ceil((q0+16)/32)

  for (int j = 0; j < nkv; ++j) {
    const int kv0 = j * 32;
    // scores S[16x32] = Q @ K^T  (K row-major == B-operand layout)
    floatx8 Sg[2];
#pragma unroll
    for (int g = 0; g < 2; ++g) {
      const u16* krow = Km + (size_t)(b * 1024 + kv0 + g * 16 + ln) * 1024 + h * 64;
      floatx8 s = zero8();
#pragma unroll
      for (int kk = 0; kk < 2; ++kk) {
        bf16x16 bk = load_frag(krow + kk * 32 + hi * 8, krow + kk * 32 + 16 + hi * 8);
        s = wmma_bf16(aq[kk], bk, s);
      }
      Sg[g] = s;
    }
    // scale 1/sqrt(64) + causal mask
#pragma unroll
    for (int g = 0; g < 2; ++g)
#pragma unroll
      for (int r = 0; r < 8; ++r) {
        int m = q0 + r + 8 * hi;
        int n = kv0 + g * 16 + ln;
        float v = Sg[g][r] * 0.125f;
        Sg[g][r] = (n > m) ? -1.0e30f : v;
      }
    // online softmax stats
    float alpha[8];
#pragma unroll
    for (int r = 0; r < 8; ++r) {
      float v = hmax16(fmaxf(Sg[0][r], Sg[1][r]));
      float Mn = fmaxf(Mr[r], v);
      alpha[r] = __expf(Mr[r] - Mn);
      Mr[r] = Mn;
    }
#pragma unroll
    for (int g = 0; g < 2; ++g)
#pragma unroll
      for (int r = 0; r < 8; ++r) Sg[g][r] = __expf(Sg[g][r] - Mr[r]);
#pragma unroll
    for (int r = 0; r < 8; ++r)
      Lr[r] = Lr[r] * alpha[r] + hsum16(Sg[0][r] + Sg[1][r]);
#pragma unroll
    for (int jb = 0; jb < 4; ++jb)
#pragma unroll
      for (int r = 0; r < 8; ++r) O[jb][r] *= alpha[r];

    // stage P (C/D layout) -> LDS row-major -> reload as A fragment
#pragma unroll
    for (int g = 0; g < 2; ++g)
#pragma unroll
      for (int r = 0; r < 8; ++r)
        Plds[wave][(r + 8 * hi) * 32 + g * 16 + ln] = f2bf(Sg[g][r]);
    asm volatile("s_wait_dscnt 0" ::: "memory");  // wave-local LDS RAW fence
    const u16* prow = &Plds[wave][ln * 32];
    bf16x16 ap = load_frag(prow + hi * 8, prow + 16 + hi * 8);

    // O += P @ V  (V^T rows are the B-operand columns)
#pragma unroll
    for (int jb = 0; jb < 4; ++jb) {
      const u16* vrow = vbase + (size_t)(jb * 16 + ln) * 1024 + kv0;
      bf16x16 bv = load_frag(vrow + hi * 8, vrow + 16 + hi * 8);
      O[jb] = wmma_bf16(ap, bv, O[jb]);
    }
  }

  // normalize and store ctx[b,s,h*64+n] (bf16)
#pragma unroll
  for (int jb = 0; jb < 4; ++jb)
#pragma unroll
    for (int r = 0; r < 8; ++r) {
      int m = q0 + r + 8 * hi;
      CTX[(size_t)(b * 1024 + m) * 1024 + h * 64 + jb * 16 + ln] =
          f2bf(O[jb][r] / Lr[r]);
    }
}

// ---------------------------------------------------------------------------
// 6. Fused residual + LayerNorm (D=1024). One block per row.
// ---------------------------------------------------------------------------
__global__ __launch_bounds__(256) void ln_kernel(const float* __restrict__ X,
                                                 const float* __restrict__ R,
                                                 const float* __restrict__ G,
                                                 const float* __restrict__ Bt,
                                                 float* __restrict__ outF,
                                                 u16* __restrict__ outB) {
  __shared__ float s1[256], s2[256];
  const int row = blockIdx.x, tid = threadIdx.x;
  const float* x = X + (size_t)row * 1024;
  const float* rr = R + (size_t)row * 1024;
  float v[4];
  float sum = 0.f, sq = 0.f;
#pragma unroll
  for (int i = 0; i < 4; ++i) {
    int d = tid + i * 256;
    v[i] = x[d] + rr[d];
    sum += v[i];
    sq += v[i] * v[i];
  }
  s1[tid] = sum; s2[tid] = sq;
  __syncthreads();
  for (int off = 128; off > 0; off >>= 1) {
    if (tid < off) { s1[tid] += s1[tid + off]; s2[tid] += s2[tid + off]; }
    __syncthreads();
  }
  float mean = s1[0] * (1.f / 1024.f);
  float var = s2[0] * (1.f / 1024.f) - mean * mean;
  float rstd = rsqrtf(var + 1e-5f);
#pragma unroll
  for (int i = 0; i < 4; ++i) {
    int d = tid + i * 256;
    float o = (v[i] - mean) * rstd * G[d] + Bt[d];
    if (outF) outF[(size_t)row * 1024 + d] = o;
    if (outB) outB[(size_t)row * 1024 + d] = f2bf(o);
  }
}

// ---------------------------------------------------------------------------
// 7. Cross-entropy: per-row NLL, then mean (deterministic, no atomics)
// ---------------------------------------------------------------------------
__global__ __launch_bounds__(256) void loss_row_kernel(const float* __restrict__ logits,
                                                       const int* __restrict__ tgt,
                                                       float* __restrict__ rowloss) {
  __shared__ float sm[256];
  const int row = blockIdx.x, tid = threadIdx.x;
  const float* lp = logits + (size_t)row * 1024;
  float mx = -3.0e38f;
  for (int d = tid; d < 1024; d += 256) mx = fmaxf(mx, lp[d]);
  sm[tid] = mx; __syncthreads();
  for (int off = 128; off > 0; off >>= 1) {
    if (tid < off) sm[tid] = fmaxf(sm[tid], sm[tid + off]);
    __syncthreads();
  }
  mx = sm[0]; __syncthreads();
  float se = 0.f;
  for (int d = tid; d < 1024; d += 256) se += __expf(lp[d] - mx);
  sm[tid] = se; __syncthreads();
  for (int off = 128; off > 0; off >>= 1) {
    if (tid < off) sm[tid] += sm[tid + off];
    __syncthreads();
  }
  if (tid == 0) rowloss[row] = (mx + logf(sm[0])) - lp[tgt[row]];
}

__global__ __launch_bounds__(256) void loss_reduce_kernel(const float* __restrict__ rowloss,
                                                          float* __restrict__ out) {
  __shared__ float sm[256];
  const int tid = threadIdx.x;
  float s = 0.f;
  for (int i = tid; i < 4096; i += 256) s += rowloss[i];
  sm[tid] = s; __syncthreads();
  for (int off = 128; off > 0; off >>= 1) {
    if (tid < off) sm[tid] += sm[tid + off];
    __syncthreads();
  }
  if (tid == 0) out[0] = sm[0] * (1.f / 4096.f);
}

// ---------------------------------------------------------------------------
// Orchestration
// ---------------------------------------------------------------------------
extern "C" void kernel_launch(void* const* d_in, const int* in_sizes, int n_in,
                              void* d_out, int out_size, void* d_ws, size_t ws_size,
                              hipStream_t stream) {
  (void)in_sizes; (void)n_in; (void)out_size; (void)ws_size;
  const int* inputs = (const int*)d_in[0];
  const int* targets = (const int*)d_in[1];
  const float* wte = (const float*)d_in[2];
  const float* wq = (const float*)d_in[3];
  const float* bq = (const float*)d_in[4];
  const float* wk = (const float*)d_in[5];
  const float* bk = (const float*)d_in[6];
  const float* wv = (const float*)d_in[7];
  const float* bv = (const float*)d_in[8];
  const float* wo = (const float*)d_in[9];
  const float* bo = (const float*)d_in[10];
  const float* w1 = (const float*)d_in[11];
  const float* b1 = (const float*)d_in[12];
  const float* w2 = (const float*)d_in[13];
  const float* b2 = (const float*)d_in[14];
  const float* ln1_g = (const float*)d_in[15];
  const float* ln1_b = (const float*)d_in[16];
  const float* ln2_g = (const float*)d_in[17];
  const float* ln2_b = (const float*)d_in[18];
  float* out = (float*)d_out;

  char* ws = (char*)d_ws;
  size_t off = 0;
  auto take = [&](size_t bytes) -> char* {
    char* p = ws + off;
    off += (bytes + 255) & ~(size_t)255;
    return p;
  };
  const size_t FMAT = (size_t)4096 * 1024 * 4;  // 16 MB f32 [4096,1024]
  const size_t BMAT = (size_t)4096 * 1024 * 2;  // 8 MB bf16 [4096,1024]

  float* xf32 = (float*)take(FMAT);
  u16* xbf = (u16*)take(BMAT);
  u16* wqb = (u16*)take((size_t)1024 * 1024 * 2);
  u16* wkb = (u16*)take((size_t)1024 * 1024 * 2);
  u16* wvb = (u16*)take((size_t)1024 * 1024 * 2);
  u16* wob = (u16*)take((size_t)1024 * 1024 * 2);
  u16* w1b = (u16*)take((size_t)1024 * 4096 * 2);
  u16* w2b = (u16*)take((size_t)4096 * 1024 * 2);
  u16* qb = (u16*)take(BMAT);
  u16* kb = (u16*)take(BMAT);
  u16* vb = (u16*)take(BMAT);
  u16* vt = (u16*)take(BMAT);
  u16* ctxb = (u16*)take(BMAT);
  float* atto = (float*)take(FMAT);
  float* adnf = (float*)take(FMAT);
  u16* adnb = (u16*)take(BMAT);
  u16* hb = (u16*)take((size_t)4096 * 4096 * 2);
  float* rowloss = (float*)take(4096 * 4);
  float* ffn = atto;  // atto is dead after LN1; reuse for FFN output

  // weight conversion
  cvt_kernel<<<512, 256, 0, stream>>>(wq, wqb, 1024 * 1024);
  cvt_kernel<<<512, 256, 0, stream>>>(wk, wkb, 1024 * 1024);
  cvt_kernel<<<512, 256, 0, stream>>>(wv, wvb, 1024 * 1024);
  cvt_kernel<<<512, 256, 0, stream>>>(wo, wob, 1024 * 1024);
  cvt_kernel<<<2048, 256, 0, stream>>>(w1, w1b, 1024 * 4096);
  cvt_kernel<<<2048, 256, 0, stream>>>(w2, w2b, 4096 * 1024);

  // embedding + PE
  embed_kernel<<<4096, 256, 0, stream>>>(inputs, wte, xf32, xbf);

  // Q,K,V projections (bf16 out)
  gemm_kernel<0, 0, 1><<<dim3(8, 32), 256, 0, stream>>>(xbf, wqb, bq, nullptr, qb, 4096, 1024, 1024);
  gemm_kernel<0, 0, 1><<<dim3(8, 32), 256, 0, stream>>>(xbf, wkb, bk, nullptr, kb, 4096, 1024, 1024);
  gemm_kernel<0, 0, 1><<<dim3(8, 32), 256, 0, stream>>>(xbf, wvb, bv, nullptr, vb, 4096, 1024, 1024);

  // V^T for P@V fragments
  vtrans_kernel<<<16384, 256, 0, stream>>>(vb, vt);

  // causal flash attention
  attn_kernel<<<dim3(8, 16, 4), 256, 0, stream>>>(qb, kb, vt, ctxb);

  // output projection (f32 out)
  gemm_kernel<0, 1, 0><<<dim3(8, 32), 256, 0, stream>>>(ctxb, wob, bo, atto, nullptr, 4096, 1024, 1024);

  // residual + LN1 (f32 + bf16 out)
  ln_kernel<<<4096, 256, 0, stream>>>(xf32, atto, ln1_g, ln1_b, adnf, adnb);

  // FFN: GELU(adn@w1+b1) (bf16 out), then @w2+b2 (f32 out)
  gemm_kernel<1, 0, 1><<<dim3(32, 32), 256, 0, stream>>>(adnb, w1b, b1, nullptr, hb, 4096, 4096, 1024);
  gemm_kernel<0, 1, 0><<<dim3(8, 32), 256, 0, stream>>>(hb, w2b, b2, ffn, nullptr, 4096, 1024, 4096);

  // residual + LN2 -> logits directly into d_out
  ln_kernel<<<4096, 256, 0, stream>>>(ffn, adnf, ln2_g, ln2_b, out, nullptr);

  // cross-entropy loss -> d_out[B*S*D]
  loss_row_kernel<<<4096, 256, 0, stream>>>(out, targets, rowloss);
  loss_reduce_kernel<<<1, 256, 0, stream>>>(rowloss, out + (size_t)4096 * 1024);
}
